// YOLOv11FactConditionedClassifierDetector_53206054863031
// MI455X (gfx1250) — compile-verified, matrix operand order checked by reading, not run on hardware
//
#include <hip/hip_runtime.h>

typedef __attribute__((ext_vector_type(16))) __bf16       v16bf;
typedef __attribute__((ext_vector_type(8)))  float        v8f;
typedef __attribute__((ext_vector_type(4)))  unsigned int u32x4;

union ABRegs { v16bf v; u32x4 u[2]; };

__device__ __forceinline__ unsigned short f2bf(float x) {
    unsigned int u = __float_as_uint(x);
    u += 0x7fffu + ((u >> 16) & 1u);           // round-to-nearest-even
    return (unsigned short)(u >> 16);
}

__device__ __forceinline__ float gelu_f(float x) {
    return 0.5f * x * (1.0f + erff(x * 0.70710678118654752f));
}

// ---------------------------------------------------------------------------
// Stage A: gb = q @ film_w + film_b    (BQ, 2C)
// ---------------------------------------------------------------------------
__global__ void film_gemm_kernel(const float* __restrict__ q,      // (BQ,768)
                                 const float* __restrict__ w,      // (768,2C)
                                 const float* __restrict__ bias,   // (2C)
                                 float* __restrict__ gb, int twoC, int total) {
    int idx = blockIdx.x * blockDim.x + threadIdx.x;
    if (idx >= total) return;
    int n  = idx % twoC;
    int bq = idx / twoC;
    const float* qr = q + (size_t)bq * 768;
    float s = bias[n];
    for (int k = 0; k < 768; ++k) s += qr[k] * w[(size_t)k * twoC + n];
    gb[idx] = s;
}

// ---------------------------------------------------------------------------
// Stage B1: fbf16(b, hw, c) = bf16(feat(b, c, hw))   -- transpose + convert
// ---------------------------------------------------------------------------
__global__ void make_fbf_kernel(const float* __restrict__ feat,
                                unsigned short* __restrict__ fbf,
                                int C, int HW, int total) {
    int idx = blockIdx.x * blockDim.x + threadIdx.x;
    if (idx >= total) return;
    int c  = idx % C;
    int hw = (idx / C) % HW;
    int b  = idx / (C * HW);
    fbf[idx] = f2bf(feat[((size_t)b * C + c) * HW + hw]);
}

// ---------------------------------------------------------------------------
// Stage B2: Wt(bq, j, c) = bf16(gamma(bq,c) * hid_w(c,j))
// ---------------------------------------------------------------------------
__global__ void make_wt_kernel(const float* __restrict__ gb,      // (BQ,2C)
                               const float* __restrict__ hidw,    // (C,128)
                               unsigned short* __restrict__ wt,   // (BQ,128,C)
                               int C, int total) {
    int idx = blockIdx.x * blockDim.x + threadIdx.x;
    if (idx >= total) return;
    int c  = idx % C;
    int j  = (idx / C) & 127;
    int bq = idx / (C * 128);
    float gamma = gb[(size_t)bq * 2 * C + c];
    wt[idx] = f2bf(gamma * hidw[(size_t)c * 128 + j]);
}

// ---------------------------------------------------------------------------
// Stage B3: biasp(bq, j) = hid_b(j) + sum_c beta(bq,c) * hid_w(c,j)
// ---------------------------------------------------------------------------
__global__ void make_biasp_kernel(const float* __restrict__ gb,
                                  const float* __restrict__ hidw,
                                  const float* __restrict__ hidb,
                                  float* __restrict__ biasp,
                                  int C, int total) {
    int idx = blockIdx.x * blockDim.x + threadIdx.x;
    if (idx >= total) return;
    int j  = idx & 127;
    int bq = idx >> 7;
    const float* beta = gb + (size_t)bq * 2 * C + C;
    float s = hidb[j];
    for (int c = 0; c < C; ++c) s += beta[c] * hidw[(size_t)c * 128 + j];
    biasp[idx] = s;
}

// ---------------------------------------------------------------------------
// Stage C: hot WMMA kernel, LDS-staged B operand.
// One workgroup (8 wave32) per (bq, block of hw tiles):
//  - stage Wt' (128xC bf16) into LDS once via global_load_async_to_lds_b128
//    (ASYNCcnt-tracked CDNA5 async copy, no VGPR roundtrip)
//  - each wave computes 16x128 h-tiles: A from global (bf16 features, L2-hot),
//    B from LDS, 8 x v_wmma_f32_16x16x32_bf16 per K-step, fp32 accumulate
//  - fused bias + exact GELU + (h @ fin_w) butterfly reduce + sigmoid -> attn
// ---------------------------------------------------------------------------
__global__ __launch_bounds__(256) void film_attn_kernel(
    const unsigned short* __restrict__ fbf,    // (B, HW, C) bf16
    const unsigned short* __restrict__ wt,     // (BQ, 128, C) bf16
    const float* __restrict__ biasp,           // (BQ, 128)
    const float* __restrict__ finw,            // (128)
    const float* __restrict__ finb,            // (1)
    float* __restrict__ attn,                  // (BQ, HW)
    int C, int HW, int Q) {
    extern __shared__ __align__(16) unsigned short sWt[];   // 128*C bf16

    const int bq   = blockIdx.y;               // 0..BQ-1
    const int b    = bq / Q;
    const int tid  = threadIdx.x;
    const int wave = tid >> 5;
    const int lane = tid & 31;
    const int n0   = lane & 15;                // row (A) / column (B,C/D) index
    const int grp  = lane >> 4;                // half-wave group
    const int kb   = grp * 8;                  // per-group K sub-offset (ISA layout)

    // ---- async stage Wt'(bq) into LDS: 16-byte chunks, ASYNCcnt tracked ----
    {
        const unsigned short* src = wt + (size_t)bq * 128 * C;
        const int nchunks = (128 * C) >> 3;    // 8 ushorts (16B) per chunk
        for (int i = tid; i < nchunks; i += 256) {
            unsigned int       loff  = (unsigned int)(unsigned long long)(sWt + i * 8);
            unsigned long long gaddr = (unsigned long long)(src + i * 8);
            asm volatile("global_load_async_to_lds_b128 %0, %1, off"
                         :: "v"(loff), "v"(gaddr) : "memory");
        }
        asm volatile("s_wait_asynccnt 0x0" ::: "memory");
        __syncthreads();
    }

    const int nk     = C >> 5;                 // K steps of 32
    const int ntiles = HW >> 4;
    const float* bp  = biasp + (size_t)bq * 128;
    const float  fb  = finb[0];

    for (int tile = blockIdx.x * 8 + wave; tile < ntiles; tile += gridDim.x * 8) {
        const unsigned short* arow = fbf + ((size_t)b * HW + (size_t)tile * 16 + n0) * C;
        const unsigned short* wbase = sWt + (size_t)n0 * C;

        v8f acc[8];
        const v8f vzero = {0.f, 0.f, 0.f, 0.f, 0.f, 0.f, 0.f, 0.f};
#pragma unroll
        for (int t = 0; t < 8; ++t) acc[t] = vzero;

        for (int kc = 0; kc < nk; ++kc) {
            const int koff = kc * 32 + kb;
            ABRegs a;
            a.u[0] = *(const u32x4*)(arow + koff);        // K = koff..koff+7
            a.u[1] = *(const u32x4*)(arow + koff + 16);   // K = koff+16..koff+23
#pragma unroll
            for (int t = 0; t < 8; ++t) {
                ABRegs bm;
                const unsigned short* brow = wbase + (size_t)(16 * t) * C + koff;
                bm.u[0] = *(const u32x4*)(brow);
                bm.u[1] = *(const u32x4*)(brow + 16);
                acc[t] = __builtin_amdgcn_wmma_f32_16x16x32_bf16(
                    false, a.v, false, bm.v, (short)0, acc[t], false, false);
            }
        }

        // Fused bias + GELU + (h @ fin_w) partial sums.
        // C/D layout: lane holds column n = 16*t + n0, rows r + 8*grp, r=0..7.
        float srow[8];
#pragma unroll
        for (int r = 0; r < 8; ++r) srow[r] = 0.f;
#pragma unroll
        for (int t = 0; t < 8; ++t) {
            const float bias = bp[16 * t + n0];
            const float fw   = finw[16 * t + n0];
#pragma unroll
            for (int r = 0; r < 8; ++r) {
                float x = acc[t][r] + bias;
                srow[r] += gelu_f(x) * fw;
            }
        }
        // Butterfly reduce across the 16 lanes of each half-wave group.
#pragma unroll
        for (int m = 1; m < 16; m <<= 1) {
#pragma unroll
            for (int r = 0; r < 8; ++r) srow[r] += __shfl_xor(srow[r], m, 32);
        }
        if (n0 == 0) {
            float* out = attn + (size_t)bq * HW + (size_t)tile * 16 + 8 * grp;
#pragma unroll
            for (int r = 0; r < 8; ++r) {
                float z = srow[r] + fb;
                out[r] = 1.0f / (1.0f + expf(-z));
            }
        }
    }
}

// ---------------------------------------------------------------------------
// Stage D: pooled(bq,c) = (gamma * sum(attn*f) + beta * sum(attn)) / (sum(attn)+eps)
// One wave per (bq, c); writes directly into the concat buffer.
// ---------------------------------------------------------------------------
__global__ __launch_bounds__(32) void pool_kernel(
    const float* __restrict__ attn,   // (BQ, HW)
    const float* __restrict__ feat,   // (B, C, HW)
    const float* __restrict__ gb,     // (BQ, 2C)
    float* __restrict__ cat,          // (BQ, 848)
    int C, int HW, int Q, int off) {
    const int bq = blockIdx.x / C;
    const int c  = blockIdx.x % C;
    const int b  = bq / Q;
    const int lane = threadIdx.x;
    const float* arow = attn + (size_t)bq * HW;
    const float* frow = feat + ((size_t)b * C + c) * HW;
    float ts = 0.f, ss = 0.f;
    for (int i = lane; i < HW; i += 32) {
        float a = arow[i];
        ss += a;
        ts += a * frow[i];
    }
#pragma unroll
    for (int m = 1; m < 32; m <<= 1) {
        ts += __shfl_xor(ts, m, 32);
        ss += __shfl_xor(ss, m, 32);
    }
    if (lane == 0) {
        float gamma = gb[(size_t)bq * 2 * C + c];
        float beta  = gb[(size_t)bq * 2 * C + C + c];
        cat[(size_t)bq * 848 + off + c] = (gamma * ts + beta * ss) / (ss + 1e-8f);
    }
}

// ---------------------------------------------------------------------------
// Stage D2: copy level-2 attn map into concat buffer columns [448, 848)
// ---------------------------------------------------------------------------
__global__ void copy_a2_kernel(const float* __restrict__ attn2,  // (BQ,400)
                               float* __restrict__ cat, int total) {
    int idx = blockIdx.x * blockDim.x + threadIdx.x;
    if (idx >= total) return;
    int i  = idx % 400;
    int bq = idx / 400;
    cat[(size_t)bq * 848 + 448 + i] = attn2[idx];
}

// ---------------------------------------------------------------------------
// Stage E: tiny fp32 MLP  848 -> 512 -> 256 -> 1
// ---------------------------------------------------------------------------
__global__ void mlp_layer_kernel(const float* __restrict__ x,    // (BQ, K)
                                 const float* __restrict__ w,    // (K, N)
                                 const float* __restrict__ bias, // (N)
                                 float* __restrict__ y,          // (BQ, N)
                                 int K, int N, int do_gelu, int total) {
    int idx = blockIdx.x * blockDim.x + threadIdx.x;
    if (idx >= total) return;
    int n  = idx % N;
    int bq = idx / N;
    const float* xr = x + (size_t)bq * K;
    float s = bias[n];
    for (int k = 0; k < K; ++k) s += xr[k] * w[(size_t)k * N + n];
    y[idx] = do_gelu ? gelu_f(s) : s;
}

// ---------------------------------------------------------------------------

extern "C" void kernel_launch(void* const* d_in, const int* in_sizes, int n_in,
                              void* d_out, int out_size, void* d_ws, size_t ws_size,
                              hipStream_t stream) {
    (void)in_sizes; (void)n_in; (void)out_size; (void)ws_size;
    const int B = 2, Q = 32, BQ = 64;
    const int Cs[3]  = {64, 128, 256};
    const int HWs[3] = {6400, 1600, 400};

    const float* feat[3] = {(const float*)d_in[0], (const float*)d_in[1], (const float*)d_in[2]};
    const float* qemb = (const float*)d_in[3];
    const float *filmw[3], *filmb[3], *hidw[3], *hidb[3], *finw[3], *finb[3];
    for (int l = 0; l < 3; ++l) {
        filmw[l] = (const float*)d_in[4 + 6 * l];
        filmb[l] = (const float*)d_in[5 + 6 * l];
        hidw[l]  = (const float*)d_in[6 + 6 * l];
        hidb[l]  = (const float*)d_in[7 + 6 * l];
        finw[l]  = (const float*)d_in[8 + 6 * l];
        finb[l]  = (const float*)d_in[9 + 6 * l];
    }
    const float* mlpw1 = (const float*)d_in[22];
    const float* mlpb1 = (const float*)d_in[23];
    const float* mlpw2 = (const float*)d_in[24];
    const float* mlpb2 = (const float*)d_in[25];
    const float* mlpw3 = (const float*)d_in[26];
    const float* mlpb3 = (const float*)d_in[27];

    // ---- carve workspace (256B-aligned regions) ----
    char* p = (char*)d_ws;
    auto carve = [&](size_t bytes) -> void* {
        void* r = (void*)p;
        p += (bytes + 255) & ~(size_t)255;
        return r;
    };
    float*          gb[3];
    float*          biasp[3];
    unsigned short* wt[3];
    unsigned short* fbf[3];
    float*          attn[3];
    for (int l = 0; l < 3; ++l) gb[l]    = (float*)carve((size_t)BQ * 2 * Cs[l] * 4);
    for (int l = 0; l < 3; ++l) biasp[l] = (float*)carve((size_t)BQ * 128 * 4);
    for (int l = 0; l < 3; ++l) wt[l]    = (unsigned short*)carve((size_t)BQ * 128 * Cs[l] * 2);
    for (int l = 0; l < 3; ++l) fbf[l]   = (unsigned short*)carve((size_t)B * HWs[l] * Cs[l] * 2);
    for (int l = 0; l < 3; ++l) attn[l]  = (float*)carve((size_t)BQ * HWs[l] * 4);
    float* cat = (float*)carve((size_t)BQ * 848 * 4);
    float* x1  = (float*)carve((size_t)BQ * 512 * 4);
    float* x2  = (float*)carve((size_t)BQ * 256 * 4);

    const int TPB = 256;
    const int catoff[3] = {0, 64, 192};

    for (int l = 0; l < 3; ++l) {
        const int C = Cs[l], HW = HWs[l];

        // A: FiLM gamma/beta GEMM
        {
            int total = BQ * 2 * C;
            film_gemm_kernel<<<(total + TPB - 1) / TPB, TPB, 0, stream>>>(
                qemb, filmw[l], filmb[l], gb[l], 2 * C, total);
        }
        // B1: bf16 transposed features
        {
            int total = B * HW * C;
            make_fbf_kernel<<<(total + TPB - 1) / TPB, TPB, 0, stream>>>(
                feat[l], fbf[l], C, HW, total);
        }
        // B2: modulated weight Wt' (bf16, transposed)
        {
            int total = BQ * 128 * C;
            make_wt_kernel<<<(total + TPB - 1) / TPB, TPB, 0, stream>>>(
                gb[l], hidw[l], wt[l], C, total);
        }
        // B3: fused bias
        {
            int total = BQ * 128;
            make_biasp_kernel<<<(total + TPB - 1) / TPB, TPB, 0, stream>>>(
                gb[l], hidw[l], hidb[l], biasp[l], C, total);
        }
        // C: WMMA GEMM (LDS-staged B) + gelu + attention head
        {
            const int ntiles = HW / 16;
            dim3 grid((ntiles + 7) / 8, BQ);
            size_t smem = (size_t)128 * C * 2;   // Wt' tile in LDS (bf16)
            film_attn_kernel<<<grid, 256, smem, stream>>>(
                fbf[l], wt[l], biasp[l], finw[l], finb[l], attn[l], C, HW, Q);
        }
        // D: attention-weighted pooling -> concat buffer
        {
            dim3 grid(BQ * C);
            pool_kernel<<<grid, 32, 0, stream>>>(
                attn[l], feat[l], gb[l], cat, C, HW, Q, catoff[l]);
        }
    }
    // D2: append level-2 attention map
    {
        int total = BQ * 400;
        copy_a2_kernel<<<(total + TPB - 1) / TPB, TPB, 0, stream>>>(attn[2], cat, total);
    }
    // E: MLP head
    {
        int total = BQ * 512;
        mlp_layer_kernel<<<(total + TPB - 1) / TPB, TPB, 0, stream>>>(
            cat, mlpw1, mlpb1, x1, 848, 512, 1, total);
    }
    {
        int total = BQ * 256;
        mlp_layer_kernel<<<(total + TPB - 1) / TPB, TPB, 0, stream>>>(
            x1, mlpw2, mlpb2, x2, 512, 256, 1, total);
    }
    {
        int total = BQ * 1;
        mlp_layer_kernel<<<1, 64, 0, stream>>>(
            x2, mlpw3, mlpb3, (float*)d_out, 256, 1, 0, total);
    }
}
